// FullModel_14963666059443
// MI455X (gfx1250) — matrix operand order repaired and checked by reading, compiled, not run on hardware
//
#include <hip/hip_runtime.h>
#include <hip/hip_bf16.h>
#include <stdint.h>

typedef __attribute__((ext_vector_type(16))) _Float16 v16h;
typedef __attribute__((ext_vector_type(8)))  float    v8f;
typedef __attribute__((ext_vector_type(4)))  int      v4i;

#define B_   4
#define T_   2048
#define D_   256
#define H_   4
#define HD_  64
#define L_   2
#define NQ_  128            // MAX_BITS * 2
#define BT_  (B_ * T_)      // 8192
#define MB_  ((size_t)1 << 20)

// ---------------------------------------------------------------------------
// gfx1250 async global->LDS path (ASYNCcnt-tracked DMA-style copies).
// Toolchain signature (from diagnostics): param0 = int4* in global AS,
// param1 = int4* in LDS AS, then two integer immediates (offset, cpol).
// Guarded so the kernel still compiles if the builtins are absent.
// ---------------------------------------------------------------------------
#if defined(__has_builtin)
#if __has_builtin(__builtin_amdgcn_global_load_async_to_lds_b128)
#define HAVE_ASYNC_LDS 1
#endif
#endif

#ifdef HAVE_ASYNC_LDS
typedef __attribute__((address_space(1))) v4i gv4i;   // global int4
typedef __attribute__((address_space(3))) v4i lv4i;   // LDS int4
static __device__ __forceinline__ void async_copy_b128(const void* g, void* l) {
  // 16 bytes per lane, global -> LDS, no VGPR data path
  __builtin_amdgcn_global_load_async_to_lds_b128((gv4i*)g, (lv4i*)l, 0, 0);
}
static __device__ __forceinline__ void wait_async0() {
#if __has_builtin(__builtin_amdgcn_s_wait_asynccnt)
  __builtin_amdgcn_s_wait_asynccnt(0);
#else
  asm volatile("s_wait_asynccnt 0x0" ::: "memory");
#endif
}
#endif

// ---------------------------------------------------------------------------
// WMMA helpers (gfx1250 wave32, 16x16x32 f16 -> f32)
// ---------------------------------------------------------------------------
static __device__ __forceinline__ v8f wmma_f16(v16h a, v16h b, v8f c) {
  // (neg_a, A, neg_b, B, c_mod, C, reuse_a, reuse_b)
  return __builtin_amdgcn_wmma_f32_16x16x32_f16(false, a, false, b, (short)0, c,
                                                false, false);
}

// A fragment 16x32 f16 from row-major storage (leading dim ld in halves).
// This code's contiguous K order (not the ISA table): lane L holds row L%16; K = kb..kb+15, kb=(L<16)?0:16
static __device__ __forceinline__ v16h frag_a(const _Float16* src, int lane, int ld) {
  int row = lane & 15;
  int kb  = (lane < 16) ? 0 : 16;
  const _Float16* p = src + row * ld + kb;
  v16h a;
#pragma unroll
  for (int i = 0; i < 8; ++i) a[i] = p[i];
#pragma unroll
  for (int i = 0; i < 8; ++i) a[8 + i] = p[8 + i];
  return a;
}

// B fragment 32x16 f16, element (k,n) stored at src[n*ld + k] (column-contig K).
// Same contiguous K order as frag_a (not the ISA table): lane L holds column L%16; K = kb..kb+15, kb=(L<16)?0:16
static __device__ __forceinline__ v16h frag_b(const _Float16* src, int lane, int ld) {
  int col = lane & 15;
  int kb  = (lane < 16) ? 0 : 16;
  const _Float16* p = src + col * ld + kb;
  v16h b;
#pragma unroll
  for (int i = 0; i < 16; ++i) b[i] = p[i];
  return b;
}

// ---------------------------------------------------------------------------
// Weight pre-transpose + convert: Wt[n*K + k] = (f16) W[k*N + n].
// ---------------------------------------------------------------------------
__global__ __launch_bounds__(256) void wconv_kernel(const float* __restrict__ W,
                                                    _Float16* __restrict__ Wt,
                                                    int K, int N) {
  int e = blockIdx.x * 256 + threadIdx.x;   // e < K*N (grid sized exactly)
  int n = e / K;
  int k = e - n * K;
  Wt[(size_t)n * K + k] = (_Float16)W[(size_t)k * N + n];
}

// ---------------------------------------------------------------------------
// Embedding gather: x[bt, d] = embed[tokens[bt], d]
// ---------------------------------------------------------------------------
__global__ __launch_bounds__(256) void embed_kernel(const int* __restrict__ tokens,
                                                    const float* __restrict__ embed,
                                                    float* __restrict__ x) {
  int bt = blockIdx.x;
  int d  = threadIdx.x;
  x[(size_t)bt * D_ + d] = embed[(size_t)tokens[bt] * D_ + d];
}

// ---------------------------------------------------------------------------
// LayerNorm over D=256, one row per 256-thread block. outH/outF optional.
// ---------------------------------------------------------------------------
__global__ __launch_bounds__(256) void ln_kernel(const float* __restrict__ x,
                                                 const float* __restrict__ w,
                                                 const float* __restrict__ b,
                                                 _Float16* __restrict__ outH,
                                                 float* __restrict__ outF) {
  __shared__ float red[256];
  int row = blockIdx.x;
  int d   = threadIdx.x;
  float v = x[(size_t)row * D_ + d];
  red[d] = v;
  __syncthreads();
  for (int s = 128; s > 0; s >>= 1) {
    if (d < s) red[d] += red[d + s];
    __syncthreads();
  }
  float mu = red[0] * (1.0f / 256.0f);
  __syncthreads();
  float c = v - mu;
  red[d] = c * c;
  __syncthreads();
  for (int s = 128; s > 0; s >>= 1) {
    if (d < s) red[d] += red[d + s];
    __syncthreads();
  }
  float var = red[0] * (1.0f / 256.0f);
  float y = c * rsqrtf(var + 1e-5f) * w[d] + b[d];
  if (outH) outH[(size_t)row * D_ + d] = (_Float16)y;
  if (outF) outF[(size_t)row * D_ + d] = y;
}

// ---------------------------------------------------------------------------
// WMMA GEMM: out = act(A_f16[MxK] @ Wt_f16[N-major, ld=K] + bias) (+ resid)
// Block: 128 threads = 4 waves; block tile 64x64; wave tile 32x32 (2x2 WMMA).
// A and Wt tiles staged via async global->LDS b128 copies (fallback: manual).
// ---------------------------------------------------------------------------
#define ACT_NONE 0
#define ACT_GELU 1

__global__ __launch_bounds__(128) void gemm_wmma_kernel(
    const _Float16* __restrict__ A, const _Float16* __restrict__ Wt,
    const float* __restrict__ bias, const float* __restrict__ resid,
    float* __restrict__ outF, _Float16* __restrict__ outH,
    int M, int N, int K, int act) {
  __shared__ __align__(16) _Float16 As[64 * 32];   // row-major, ld 32
  __shared__ __align__(16) _Float16 Bs[64 * 32];   // Wt tile: [n][k], ld 32

  int tid  = threadIdx.x;
  int lane = tid & 31;
  int wave = tid >> 5;
  int m0 = blockIdx.x * 64;
  int n0 = blockIdx.y * 64;
  int wr = (wave & 1) * 32;
  int wc = (wave >> 1) * 32;

  v8f c00 = {}, c01 = {}, c10 = {}, c11 = {};

  for (int k0 = 0; k0 < K; k0 += 32) {
    __syncthreads();   // previous iteration's fragment reads are done
#ifdef HAVE_ASYNC_LDS
    // Each tile is 64 rows x 32 halves = 4 KB = 256 16-byte chunks.
    // 128 threads x 2 chunks each, for A and for B: pure LDS DMA, no VGPR data.
#pragma unroll
    for (int u = 0; u < 2; ++u) {
      int c   = tid * 2 + u;        // 0..255
      int r   = c >> 2;             // row 0..63
      int off = (c & 3) * 8;        // half offset within row
      async_copy_b128(A  + (size_t)(m0 + r) * K + k0 + off, As + r * 32 + off);
      async_copy_b128(Wt + (size_t)(n0 + r) * K + k0 + off, Bs + r * 32 + off);
    }
    wait_async0();
#else
    for (int i = tid; i < 64 * 16; i += 128) {
      int r = i >> 4, cidx = i & 15;
      ((uint32_t*)As)[r * 16 + cidx] =
          ((const uint32_t*)A)[(((size_t)(m0 + r) * K + k0) >> 1) + cidx];
      ((uint32_t*)Bs)[r * 16 + cidx] =
          ((const uint32_t*)Wt)[(((size_t)(n0 + r) * K + k0) >> 1) + cidx];
    }
#endif
    __syncthreads();

    v16h a0 = frag_a(As + (wr + 0) * 32, lane, 32);
    v16h a1 = frag_a(As + (wr + 16) * 32, lane, 32);
    v16h b0 = frag_b(Bs + (wc + 0) * 32, lane, 32);
    v16h b1 = frag_b(Bs + (wc + 16) * 32, lane, 32);
    c00 = wmma_f16(a0, b0, c00);
    c01 = wmma_f16(a0, b1, c01);
    c10 = wmma_f16(a1, b0, c10);
    c11 = wmma_f16(a1, b1, c11);
  }

  // Epilogue per C layout: VGPR r -> row r + (lane<16?0:8), col lane&15
  int col = lane & 15;
  int rhi = (lane < 16) ? 0 : 8;
#pragma unroll
  for (int ti = 0; ti < 2; ++ti) {
#pragma unroll
    for (int tj = 0; tj < 2; ++tj) {
      v8f cc = (ti == 0) ? ((tj == 0) ? c00 : c01) : ((tj == 0) ? c10 : c11);
#pragma unroll
      for (int r = 0; r < 8; ++r) {
        int gm = m0 + wr + ti * 16 + rhi + r;
        int gn = n0 + wc + tj * 16 + col;
        float v = cc[r] + bias[gn];
        if (act == ACT_GELU) v = 0.5f * v * (1.0f + erff(v * 0.70710678118654752f));
        if (resid) v += resid[(size_t)gm * N + gn];
        if (outF) outF[(size_t)gm * N + gn] = v;
        if (outH) outH[(size_t)gm * N + gn] = (_Float16)v;
      }
    }
  }
}

// ---------------------------------------------------------------------------
// RoPE + head split: qkv f32 (B,T,3,H,HD) -> qf/kf/vf f16 (B,H,T,HD)
// ---------------------------------------------------------------------------
__global__ __launch_bounds__(256) void rope_kernel(const float* __restrict__ qkv,
                                                   _Float16* __restrict__ qf,
                                                   _Float16* __restrict__ kf,
                                                   _Float16* __restrict__ vf) {
  int bt = blockIdx.x;
  int b = bt >> 11;
  int t = bt & (T_ - 1);
  int tid = threadIdx.x;
  int h  = tid >> 6;
  int hd = tid & 63;
  size_t base = (size_t)bt * 3 * D_;
  float qv = qkv[base + 0 * D_ + h * HD_ + hd];
  float kv = qkv[base + 1 * D_ + h * HD_ + hd];
  float vv = qkv[base + 2 * D_ + h * HD_ + hd];
  int fi = hd & 31;
  float ang = (float)t * __expf(-(float)fi * (9.210340371976184f / 32.0f));
  float cs = __cosf(ang), sn = __sinf(ang);
  int partner = (hd < 32) ? hd + 32 : hd - 32;
  float sgn = (hd < 32) ? -1.0f : 1.0f;
  float qp = qkv[base + 0 * D_ + h * HD_ + partner];
  float kp = qkv[base + 1 * D_ + h * HD_ + partner];
  float qo = qv * cs + sgn * qp * sn;
  float ko = kv * cs + sgn * kp * sn;
  size_t ob = ((size_t)((b * H_ + h) * T_ + t)) * HD_ + hd;
  qf[ob] = (_Float16)qo;
  kf[ob] = (_Float16)ko;
  vf[ob] = (_Float16)vv;
}

// ---------------------------------------------------------------------------
// Flash-style causal attention, one wave32 per (b, h, 16-query tile).
// K/V chunks async-staged into LDS; S = Q@K^T and O += P@V via WMMA with
// online softmax through a 16x32 LDS crossover tile.
// ---------------------------------------------------------------------------
__global__ __launch_bounds__(32) void attn_kernel(const _Float16* __restrict__ qf,
                                                  const _Float16* __restrict__ kf,
                                                  const _Float16* __restrict__ vf,
                                                  _Float16* __restrict__ outH) {
  __shared__ float    Ss[16 * 32];
  __shared__ _Float16 Ps[16 * 32];
  __shared__ float    m_s[16], l_s[16], scale_s[16];
#ifdef HAVE_ASYNC_LDS
  __shared__ __align__(16) _Float16 Ks[32 * 64];   // 4 KB key chunk
  __shared__ __align__(16) _Float16 Vs[32 * 64];   // 4 KB value chunk
#endif

  int wg = blockIdx.x;              // B*H*(T/16) = 2048
  int qt = wg & 127;
  int h  = (wg >> 7) & (H_ - 1);
  int b  = wg >> 9;
  int lane = threadIdx.x & 31;
  int q0 = qt * 16;

  const size_t headbase = (size_t)(b * H_ + h) * T_ * HD_;
  const _Float16* qbase = qf + headbase + (size_t)q0 * HD_;

  // Q tile A-fragments for the two hd chunks (K=0..31, 32..63)
  v16h aq[2];
  {
    int row = lane & 15;
    int kb  = (lane < 16) ? 0 : 16;
#pragma unroll
    for (int c = 0; c < 2; ++c) {
      const _Float16* p = qbase + row * HD_ + c * 32 + kb;
#pragma unroll
      for (int i = 0; i < 8; ++i) aq[c][i] = p[i];
#pragma unroll
      for (int i = 0; i < 8; ++i) aq[c][8 + i] = p[8 + i];
    }
  }

  if (lane < 16) { m_s[lane] = -1e30f; l_s[lane] = 0.0f; }
  v8f o[4] = {};
  __syncthreads();

  int col = lane & 15;
  int rhi = (lane < 16) ? 0 : 8;

  for (int kt0 = 0; kt0 <= q0 + 15; kt0 += 32) {
    const _Float16* kbp = kf + headbase + (size_t)kt0 * HD_;
    const _Float16* vbp = vf + headbase + (size_t)kt0 * HD_;
    const _Float16* ksrc;
    const _Float16* vsrc;
#ifdef HAVE_ASYNC_LDS
    // Stage 32x64-half K and V chunks: 256 16B chunks each, 8 per lane.
#pragma unroll
    for (int i = 0; i < 8; ++i) {
      int c   = i * 32 + lane;
      int r   = c >> 3;
      int off = (c & 7) * 8;
      async_copy_b128(kbp + r * HD_ + off, Ks + r * HD_ + off);
      async_copy_b128(vbp + r * HD_ + off, Vs + r * HD_ + off);
    }
    wait_async0();
    __syncthreads();
    ksrc = Ks;
    vsrc = Vs;
#else
    ksrc = kbp;
    vsrc = vbp;
#endif

    // ---- S = Q @ K^T for 32 keys (two 16-key C fragments) ----
#pragma unroll
    for (int cf = 0; cf < 2; ++cf) {
      v8f sc = {};
#pragma unroll
      for (int kc = 0; kc < 2; ++kc) {
        v16h bk;
        int kb = (lane < 16) ? 0 : 16;
        const _Float16* p = ksrc + (cf * 16 + col) * HD_ + kc * 32 + kb;
#pragma unroll
        for (int i = 0; i < 16; ++i) bk[i] = p[i];
        sc = wmma_f16(aq[kc], bk, sc);
      }
#pragma unroll
      for (int r = 0; r < 8; ++r) Ss[(rhi + r) * 32 + cf * 16 + col] = sc[r];
    }
    __syncthreads();

    // ---- Online softmax: lane m < 16 owns query row m ----
    if (lane < 16) {
      int m = lane;
      int t = q0 + m;
      float mo = m_s[m];
      float sv[32];
      float rmax = -1e30f;
#pragma unroll
      for (int j = 0; j < 32; ++j) {
        int key = kt0 + j;
        float v = Ss[m * 32 + j] * 0.125f;          // 1/sqrt(64)
        if (key > t) v = -1e30f;                    // causal mask
        sv[j] = v;
        rmax = fmaxf(rmax, v);
      }
      float mn  = fmaxf(mo, rmax);
      float scl = __expf(mo - mn);
      float ls  = l_s[m] * scl;
#pragma unroll
      for (int j = 0; j < 32; ++j) {
        float p = __expf(sv[j] - mn);
        ls += p;
        Ps[m * 32 + j] = (_Float16)p;
      }
      m_s[m] = mn; l_s[m] = ls; scale_s[m] = scl;
    }
    __syncthreads();

    // ---- Rescale O by exp(m_old - m_new) per row ----
#pragma unroll
    for (int n = 0; n < 4; ++n)
#pragma unroll
      for (int r = 0; r < 8; ++r) o[n][r] *= scale_s[rhi + r];

    // ---- O += P @ V ----
    v16h pA = frag_a(Ps, lane, 32);
    int kb = (lane < 16) ? 0 : 16;
#pragma unroll
    for (int n = 0; n < 4; ++n) {
      v16h bv;
#pragma unroll
      for (int i = 0; i < 16; ++i) bv[i] = vsrc[(kb + i) * HD_ + n * 16 + col];
      o[n] = wmma_f16(pA, bv, o[n]);
    }
    __syncthreads();   // protect Ss/Ps/Ks/Vs for next chunk
  }

  // ---- Finalize: divide by row sum, write (b, t, h*64 + hd) f16 ----
#pragma unroll
  for (int n = 0; n < 4; ++n)
#pragma unroll
    for (int r = 0; r < 8; ++r) {
      int m = rhi + r;
      int t = q0 + m;
      float val = o[n][r] / l_s[m];
      outH[((size_t)(b * T_ + t)) * D_ + h * HD_ + n * 16 + col] = (_Float16)val;
    }
}

// ---------------------------------------------------------------------------
// Pooled-query softmax: q_attn[b,q,t] = softmax_t(dot(query_q, x_t)/16)
// ---------------------------------------------------------------------------
__global__ __launch_bounds__(256) void qscore_kernel(const float* __restrict__ queries,
                                                     const float* __restrict__ xf,
                                                     float* __restrict__ qattn) {
  __shared__ float red[256];
  int bq = blockIdx.x;
  int b  = bq >> 7;
  int tid = threadIdx.x;
  const float* qv = queries + (size_t)(bq & 127) * D_;

  float sc[8];
#pragma unroll
  for (int i = 0; i < 8; ++i) {
    int t = tid + i * 256;
    const float* xr = xf + (size_t)(b * T_ + t) * D_;
    float s = 0.0f;
    for (int d = 0; d < D_; ++d) s += qv[d] * xr[d];
    sc[i] = s * 0.0625f;   // 1/sqrt(256)
  }
  float mx = sc[0];
#pragma unroll
  for (int i = 1; i < 8; ++i) mx = fmaxf(mx, sc[i]);
  red[tid] = mx;
  __syncthreads();
  for (int s = 128; s > 0; s >>= 1) {
    if (tid < s) red[tid] = fmaxf(red[tid], red[tid + s]);
    __syncthreads();
  }
  float gmax = red[0];
  __syncthreads();
  float ssum = 0.0f;
#pragma unroll
  for (int i = 0; i < 8; ++i) { sc[i] = __expf(sc[i] - gmax); ssum += sc[i]; }
  red[tid] = ssum;
  __syncthreads();
  for (int s = 128; s > 0; s >>= 1) {
    if (tid < s) red[tid] += red[tid + s];
    __syncthreads();
  }
  float inv = 1.0f / red[0];
#pragma unroll
  for (int i = 0; i < 8; ++i)
    qattn[(size_t)bq * T_ + tid + i * 256] = sc[i] * inv;
}

// selected[b,q,d] = sum_t q_attn[b,q,t] * x[b,t,d]
__global__ __launch_bounds__(256) void selected_kernel(const float* __restrict__ qattn,
                                                       const float* __restrict__ xf,
                                                       float* __restrict__ sel) {
  int bq = blockIdx.x;
  int b  = bq >> 7;
  int d  = threadIdx.x;
  const float* ar = qattn + (size_t)bq * T_;
  float s = 0.0f;
  for (int t = 0; t < T_; ++t) s += ar[t] * xf[(size_t)(b * T_ + t) * D_ + d];
  sel[(size_t)bq * D_ + d] = s;
}

// bits[bq] = sigmoid(dot(selected[bq,:], outp_w) + outp_b)
__global__ __launch_bounds__(256) void bits_kernel(const float* __restrict__ sel,
                                                   const float* __restrict__ w,
                                                   const float* __restrict__ bb,
                                                   float* __restrict__ pairs_out) {
  __shared__ float red[256];
  int bq = blockIdx.x;
  int d  = threadIdx.x;
  red[d] = sel[(size_t)bq * D_ + d] * w[d];
  __syncthreads();
  for (int s = 128; s > 0; s >>= 1) {
    if (d < s) red[d] += red[d + s];
    __syncthreads();
  }
  if (d == 0) pairs_out[bq] = 1.0f / (1.0f + __expf(-(red[0] + bb[0])));
}

// Sequential 64-step adder MLP: 4 chains x 64 hidden, one block of 256 threads.
__global__ __launch_bounds__(256) void adder_kernel(const float* __restrict__ pairs,
                                                    const float* __restrict__ w1,
                                                    const float* __restrict__ b1,
                                                    const float* __restrict__ w2,
                                                    const float* __restrict__ b2,
                                                    const float* __restrict__ w3,
                                                    const float* __restrict__ b3,
                                                    float* __restrict__ sums) {
  __shared__ float h1[4][64], h2[4][64], carry_s[4], o_s[4][2];
  int tid = threadIdx.x;
  int chain = tid >> 6;
  int j = tid & 63;
  if (j == 0) carry_s[chain] = 0.0f;
  __syncthreads();
  for (int step = 0; step < 64; ++step) {
    float a  = pairs[chain * 128 + step * 2 + 0];
    float bb = pairs[chain * 128 + step * 2 + 1];
    float c  = carry_s[chain];
    float v = a * w1[j] + bb * w1[64 + j] + c * w1[128 + j] + b1[j];
    h1[chain][j] = fmaxf(v, 0.0f);
    __syncthreads();
    float s = b2[j];
    for (int k = 0; k < 64; ++k) s += h1[chain][k] * w2[k * 64 + j];
    h2[chain][j] = fmaxf(s, 0.0f);
    __syncthreads();
    if (j < 2) {
      float s2 = b3[j];
      for (int k = 0; k < 64; ++k) s2 += h2[chain][k] * w3[k * 2 + j];
      o_s[chain][j] = 1.0f / (1.0f + __expf(-s2));
    }
    __syncthreads();
    if (j == 0) {
      sums[chain * 65 + step] = o_s[chain][0];
      carry_s[chain] = o_s[chain][1];
    }
    __syncthreads();
  }
  if (tid < 4) sums[tid * 65 + 64] = carry_s[tid];
}

// ---------------------------------------------------------------------------
// Host-side orchestration
// ---------------------------------------------------------------------------
extern "C" void kernel_launch(void* const* d_in, const int* in_sizes, int n_in,
                              void* d_out, int out_size, void* d_ws, size_t ws_size,
                              hipStream_t stream) {
  const int*   tokens  = (const int*)d_in[0];
  const float* embed   = (const float*)d_in[1];
  const float* ln1_w   = (const float*)d_in[2];
  const float* ln1_b   = (const float*)d_in[3];
  const float* qkv_w   = (const float*)d_in[4];
  const float* qkv_b   = (const float*)d_in[5];
  const float* proj_w  = (const float*)d_in[6];
  const float* proj_b  = (const float*)d_in[7];
  const float* ln2_w   = (const float*)d_in[8];
  const float* ln2_b   = (const float*)d_in[9];
  const float* ffn1_w  = (const float*)d_in[10];
  const float* ffn1_b  = (const float*)d_in[11];
  const float* ffn2_w  = (const float*)d_in[12];
  const float* ffn2_b  = (const float*)d_in[13];
  const float* lnf_w   = (const float*)d_in[14];
  const float* lnf_b   = (const float*)d_in[15];
  const float* queries = (const float*)d_in[16];
  const float* outp_w  = (const float*)d_in[17];
  const float* outp_b  = (const float*)d_in[18];
  const float* mlp_w1  = (const float*)d_in[19];
  const float* mlp_b1  = (const float*)d_in[20];
  const float* mlp_w2  = (const float*)d_in[21];
  const float* mlp_b2  = (const float*)d_in[22];
  const float* mlp_w3  = (const float*)d_in[23];
  const float* mlp_b3  = (const float*)d_in[24];

  // Workspace layout (all MiB-aligned)
  char* ws = (char*)d_ws;
  float*    x       = (float*)(ws + 0 * MB_);     //  8 MiB residual stream f32
  _Float16* xn      = (_Float16*)(ws + 8 * MB_);  //  4 MiB LN out f16
  float*    qkvbuf  = (float*)(ws + 12 * MB_);    // 24 MiB QKV GEMM out f32
  _Float16* qfb     = (_Float16*)(ws + 36 * MB_); //  4 MiB
  _Float16* kfb     = (_Float16*)(ws + 40 * MB_); //  4 MiB
  _Float16* vfb     = (_Float16*)(ws + 44 * MB_); //  4 MiB
  _Float16* attnout = (_Float16*)(ws + 48 * MB_); //  4 MiB
  _Float16* hbuf    = (_Float16*)(ws + 52 * MB_); // 16 MiB FFN hidden f16
  float*    xf      = (float*)(ws + 68 * MB_);    //  8 MiB final LN f32
  float*    sel     = (float*)(ws + 76 * MB_);    //  0.5 MiB
  // f16 transposed weights (~3 MiB total)
  _Float16* qkvw_t  = (_Float16*)(ws + 80 * MB_);          // 2 x 196608
  _Float16* projw_t = qkvw_t  + (size_t)L_ * D_ * 3 * D_;  // 2 x 65536
  _Float16* ffn1w_t = projw_t + (size_t)L_ * D_ * D_;      // 2 x 262144
  _Float16* ffn2w_t = ffn1w_t + (size_t)L_ * D_ * 4 * D_;  // 2 x 262144

  float* out        = (float*)d_out;
  float* out_sums   = out;                        // (4, 65)
  float* out_pairs  = out + 260;                  // (4, 64, 2) == bits (4,128)
  float* out_qattn  = out + 772;                  // (4, 128, 2048)

  // One-time weight convert/transpose to f16 N-major (ld = K)
  for (int l = 0; l < L_; ++l) {
    wconv_kernel<<<(D_ * 3 * D_) / 256, 256, 0, stream>>>(
        qkv_w + (size_t)l * D_ * 3 * D_, qkvw_t + (size_t)l * D_ * 3 * D_, D_, 3 * D_);
    wconv_kernel<<<(D_ * D_) / 256, 256, 0, stream>>>(
        proj_w + (size_t)l * D_ * D_, projw_t + (size_t)l * D_ * D_, D_, D_);
    wconv_kernel<<<(D_ * 4 * D_) / 256, 256, 0, stream>>>(
        ffn1_w + (size_t)l * D_ * 4 * D_, ffn1w_t + (size_t)l * D_ * 4 * D_, D_, 4 * D_);
    wconv_kernel<<<(4 * D_ * D_) / 256, 256, 0, stream>>>(
        ffn2_w + (size_t)l * 4 * D_ * D_, ffn2w_t + (size_t)l * 4 * D_ * D_, 4 * D_, D_);
  }

  embed_kernel<<<BT_, 256, 0, stream>>>(tokens, embed, x);

  for (int l = 0; l < L_; ++l) {
    ln_kernel<<<BT_, 256, 0, stream>>>(x, ln1_w + l * D_, ln1_b + l * D_, xn, nullptr);
    gemm_wmma_kernel<<<dim3(BT_ / 64, (3 * D_) / 64), 128, 0, stream>>>(
        xn, qkvw_t + (size_t)l * D_ * 3 * D_, qkv_b + (size_t)l * 3 * D_,
        nullptr, qkvbuf, nullptr, BT_, 3 * D_, D_, ACT_NONE);
    rope_kernel<<<BT_, 256, 0, stream>>>(qkvbuf, qfb, kfb, vfb);
    attn_kernel<<<B_ * H_ * (T_ / 16), 32, 0, stream>>>(qfb, kfb, vfb, attnout);
    gemm_wmma_kernel<<<dim3(BT_ / 64, D_ / 64), 128, 0, stream>>>(
        attnout, projw_t + (size_t)l * D_ * D_, proj_b + (size_t)l * D_,
        x, x, nullptr, BT_, D_, D_, ACT_NONE);
    ln_kernel<<<BT_, 256, 0, stream>>>(x, ln2_w + l * D_, ln2_b + l * D_, xn, nullptr);
    gemm_wmma_kernel<<<dim3(BT_ / 64, (4 * D_) / 64), 128, 0, stream>>>(
        xn, ffn1w_t + (size_t)l * D_ * 4 * D_, ffn1_b + (size_t)l * 4 * D_,
        nullptr, nullptr, hbuf, BT_, 4 * D_, D_, ACT_GELU);
    gemm_wmma_kernel<<<dim3(BT_ / 64, D_ / 64), 128, 0, stream>>>(
        hbuf, ffn2w_t + (size_t)l * 4 * D_ * D_, ffn2_b + (size_t)l * D_,
        x, x, nullptr, BT_, D_, 4 * D_, ACT_NONE);
  }

  ln_kernel<<<BT_, 256, 0, stream>>>(x, lnf_w, lnf_b, nullptr, xf);
  qscore_kernel<<<B_ * NQ_, 256, 0, stream>>>(queries, xf, out_qattn);
  selected_kernel<<<B_ * NQ_, 256, 0, stream>>>(out_qattn, xf, sel);
  bits_kernel<<<B_ * NQ_, 256, 0, stream>>>(sel, outp_w, outp_b, out_pairs);
  adder_kernel<<<1, 256, 0, stream>>>(out_pairs, mlp_w1, mlp_b1, mlp_w2, mlp_b2,
                                      mlp_w3, mlp_b3, out_sums);
}
